// EnhancedMambaBlock_3006477107873
// MI455X (gfx1250) — matrix operand order repaired and checked by reading, not compiled, not run on hardware
//
#include <hip/hip_runtime.h>
#include <hip/hip_bf16.h>
#include <stdint.h>

#define D_MODEL 256
#define D_INNER 512
#define D_STATE 16
#define DT_RANK 16
#define NBATCH  4
#define SEQ     4096
#define MTOK    (NBATCH * SEQ)   // 16384 tokens

typedef __attribute__((ext_vector_type(16))) _Float16 v16h;
typedef __attribute__((ext_vector_type(8)))  _Float16 v8h;
typedef __attribute__((ext_vector_type(8)))  float    v8f;

// Load a 16-half WMMA operand fragment as two 16-byte chunks.
// A-fragments: halves {0..7} at p, {8..15} at p+16  (K split 0-7/16-23 per lane group)
// B-fragments: same split as A (second chunk at p+16)
__device__ __forceinline__ v16h ld_frag(const _Float16* __restrict__ p, const int gap)
{
    v8h lo = *(const v8h*)p;
    v8h hi = *(const v8h*)(p + gap);
    v16h r;
#pragma unroll
    for (int i = 0; i < 8; i++) { r[i] = lo[i]; r[i + 8] = hi[i]; }
    return r;
}

// ---------------------------------------------------------------------------
// f16-in / f32-accumulate WMMA GEMM:  out[M,N] = A[M,K] @ W[N,K]^T
// Each wave computes a (16*MT) x (16*NT) strip with MT*NT accumulators.
// K-loop is software-pipelined (ping-pong fragment buffers) so VMEM loads for
// step k+32 overlap the WMMAs of step k.  Requires K % 64 == 0.
// ---------------------------------------------------------------------------
#define EPI_F32     0   // store f32
#define EPI_F16     1   // store f16 (with column offset into wider buffer)
#define EPI_BIASRES 2   // out = acc + bias[n] + resid[m,n] -> f32
#define EPI_SILU16  3   // out = silu(acc + bias[n]) -> f16

template <int EPI, int MT, int NT>
__global__ __launch_bounds__(256)
void wmma_gemm_kernel(const _Float16* __restrict__ A,
                      const _Float16* __restrict__ Bw,
                      int M, int Nn, int K,
                      float* __restrict__ outf, int ldo,
                      _Float16* __restrict__ outh, int ldh, int coloff,
                      const float* __restrict__ bias,
                      const float* __restrict__ resid, int ldr)
{
    const int lane = threadIdx.x & 31;
    const int wid  = (blockIdx.x * blockDim.x + threadIdx.x) >> 5;
    const int ngroups = (Nn / 16) / NT;
    const int totalW  = (M / (16 * MT)) * ngroups;
    if (wid >= totalW) return;
    const int mt0 = (wid / ngroups) * MT;
    const int n0  = (wid % ngroups) * (NT * 16);

    // A-fragment (16x32 f16, wave32): lanes 0-15/16-31 hold rows M=lane&15;
    //   lane<16: halves -> K {k0..k0+7, k0+16..k0+23}; lane>=16: +8 / +24.
    const int rowA = lane & 15;
    const int hiA  = (lane >> 4) * 8;
    // B-fragment (32x16 f16): lane = column (mod 16);
    //   same K order as A: lane<16: {k0..k0+7, k0+16..k0+23}; lane>=16: +8 / +24.
    const int colB = lane & 15;
    const int hiB  = (lane >> 4) * 8;

    const _Float16* aBase[MT];
#pragma unroll
    for (int mi = 0; mi < MT; mi++)
        aBase[mi] = A + (size_t)((mt0 + mi) * 16 + rowA) * K + hiA;
    const _Float16* bBase[NT];
#pragma unroll
    for (int nt = 0; nt < NT; nt++)
        bBase[nt] = Bw + (size_t)(n0 + nt * 16 + colB) * K + hiB;

    v8f acc[MT][NT] = {};

    // prologue: fill buffer 0 with K-step 0
    v16h a0[MT], b0[NT], a1[MT], b1[NT];
#pragma unroll
    for (int mi = 0; mi < MT; mi++) a0[mi] = ld_frag(aBase[mi], 16);
#pragma unroll
    for (int nt = 0; nt < NT; nt++) b0[nt] = ld_frag(bBase[nt], 16);

    for (int k = 0; k < K; k += 64) {
        // load K-step k+32 into buffer 1 (always exists: K % 64 == 0)
#pragma unroll
        for (int mi = 0; mi < MT; mi++) a1[mi] = ld_frag(aBase[mi] + k + 32, 16);
#pragma unroll
        for (int nt = 0; nt < NT; nt++) b1[nt] = ld_frag(bBase[nt] + k + 32, 16);
        if (k + 128 < K) __builtin_prefetch(aBase[0] + k + 128, 0, 1);

        // compute K-step k from buffer 0
#pragma unroll
        for (int mi = 0; mi < MT; mi++)
#pragma unroll
            for (int nt = 0; nt < NT; nt++)
                acc[mi][nt] = __builtin_amdgcn_wmma_f32_16x16x32_f16(
                    false, a0[mi], false, b0[nt], (short)0, acc[mi][nt], false, false);

        // load K-step k+64 into buffer 0 (overlaps buffer-1 WMMAs)
        if (k + 64 < K) {
#pragma unroll
            for (int mi = 0; mi < MT; mi++) a0[mi] = ld_frag(aBase[mi] + k + 64, 16);
#pragma unroll
            for (int nt = 0; nt < NT; nt++) b0[nt] = ld_frag(bBase[nt] + k + 64, 16);
        }

        // compute K-step k+32 from buffer 1
#pragma unroll
        for (int mi = 0; mi < MT; mi++)
#pragma unroll
            for (int nt = 0; nt < NT; nt++)
                acc[mi][nt] = __builtin_amdgcn_wmma_f32_16x16x32_f16(
                    false, a1[mi], false, b1[nt], (short)0, acc[mi][nt], false, false);
    }

    // D layout (16x16 f32): VGPR r, lane l -> (M = r + 8*(l>=16), N = l&15)
    const int colD = lane & 15;
#pragma unroll
    for (int mi = 0; mi < MT; mi++) {
        const int rowBase = (mt0 + mi) * 16 + (lane >> 4) * 8;
#pragma unroll
        for (int nt = 0; nt < NT; nt++) {
            const int col = n0 + nt * 16 + colD;
#pragma unroll
            for (int r = 0; r < 8; r++) {
                const int row = rowBase + r;
                float v = acc[mi][nt][r];
                if constexpr (EPI == EPI_F32) {
                    outf[(size_t)row * ldo + col] = v;
                } else if constexpr (EPI == EPI_F16) {
                    outh[(size_t)row * ldh + coloff + col] = (_Float16)v;
                } else if constexpr (EPI == EPI_BIASRES) {
                    v += bias[col] + resid[(size_t)row * ldr + col];
                    outf[(size_t)row * ldo + col] = v;
                } else { // EPI_SILU16
                    v += bias[col];
                    v = v * (1.f / (1.f + __expf(-v)));
                    outh[(size_t)row * ldh + col] = (_Float16)v;
                }
            }
        }
    }
}

// ---------------------------------------------------------------------------
// LayerNorm over 256 features, one wave32 per row, emits f16 for WMMA input.
// ---------------------------------------------------------------------------
__global__ __launch_bounds__(256)
void ln_f16_kernel(const float* __restrict__ x, const float* __restrict__ g,
                   const float* __restrict__ b, _Float16* __restrict__ out, int rows)
{
    const int row = (blockIdx.x * blockDim.x + threadIdx.x) >> 5;
    if (row >= rows) return;
    const int lane = threadIdx.x & 31;
    const float* xr = x + (size_t)row * D_MODEL;
    float v[8];
    float s = 0.f;
#pragma unroll
    for (int i = 0; i < 8; i++) { v[i] = xr[lane + 32 * i]; s += v[i]; }
#pragma unroll
    for (int o = 16; o > 0; o >>= 1) s += __shfl_xor(s, o, 32);
    const float mu = s * (1.f / D_MODEL);
    float q = 0.f;
#pragma unroll
    for (int i = 0; i < 8; i++) { float d = v[i] - mu; q += d * d; }
#pragma unroll
    for (int o = 16; o > 0; o >>= 1) q += __shfl_xor(q, o, 32);
    const float inv = rsqrtf(q * (1.f / D_MODEL) + 1e-5f);
#pragma unroll
    for (int i = 0; i < 8; i++) {
        int c = lane + 32 * i;
        out[(size_t)row * D_MODEL + c] = (_Float16)((v[i] - mu) * inv * g[c] + b[c]);
    }
}

// ---------------------------------------------------------------------------
// Depthwise causal conv (width 4) + SiLU. reverse=1 implements
// flip -> conv -> flip, i.e. xc[t] = sum_j w[j] * xs[t+3-j] with right pad.
// xs is the first D_INNER columns of xz[M,1024].
// ---------------------------------------------------------------------------
__global__ __launch_bounds__(256)
void conv_silu_kernel(const float* __restrict__ xz, const float* __restrict__ cw,
                      const float* __restrict__ cb, float* __restrict__ xsact,
                      _Float16* __restrict__ xsact16, int reverse)
{
    const int idx = blockIdx.x * blockDim.x + threadIdx.x;
    if (idx >= MTOK * D_INNER) return;
    const int d  = idx & (D_INNER - 1);
    const int m  = idx >> 9;
    const int bb = m >> 12;           // batch (N = 4096 = 2^12)
    const int t  = m & (SEQ - 1);
    float acc = cb[d];
    const float w0 = cw[d * 4 + 0], w1 = cw[d * 4 + 1];
    const float w2 = cw[d * 4 + 2], w3 = cw[d * 4 + 3];
    const size_t base = (size_t)((size_t)bb << 12) * 1024 + d;
    if (!reverse) {
        if (t >= 3) acc += w0 * xz[base + (size_t)(t - 3) * 1024];
        if (t >= 2) acc += w1 * xz[base + (size_t)(t - 2) * 1024];
        if (t >= 1) acc += w2 * xz[base + (size_t)(t - 1) * 1024];
        acc += w3 * xz[base + (size_t)t * 1024];
    } else {
        acc += w3 * xz[base + (size_t)t * 1024];
        if (t + 1 < SEQ) acc += w2 * xz[base + (size_t)(t + 1) * 1024];
        if (t + 2 < SEQ) acc += w1 * xz[base + (size_t)(t + 2) * 1024];
        if (t + 3 < SEQ) acc += w0 * xz[base + (size_t)(t + 3) * 1024];
    }
    const float sv = acc * (1.f / (1.f + __expf(-acc)));
    xsact[idx]   = sv;
    xsact16[idx] = (_Float16)sv;
}

// ---------------------------------------------------------------------------
// dt = softplus(xdbl[:, :16] @ wdt^T + bdt)   (K=16: VALU, fused softplus)
// ---------------------------------------------------------------------------
__global__ __launch_bounds__(256)
void dtproj_kernel(const float* __restrict__ xdbl, const float* __restrict__ wdt,
                   const float* __restrict__ bdt, float* __restrict__ dt)
{
    const int idx = blockIdx.x * blockDim.x + threadIdx.x;
    if (idx >= MTOK * D_INNER) return;
    const int d = idx & (D_INNER - 1);
    const int m = idx >> 9;
    const float* r = xdbl + (size_t)m * 48;
    const float* w = wdt + d * 16;
    float acc = bdt[d];
#pragma unroll
    for (int i = 0; i < 16; i++) acc += r[i] * w[i];
    dt[idx] = (acc > 20.f) ? acc : log1pf(__expf(acc));
}

// ---------------------------------------------------------------------------
// Selective scan: one thread per (batch, channel); 16 states in registers.
// Fused epilogue: y = scan + x*dskip, gated with silu(z), stored as f16
// ready for the out_proj WMMA. reverse=1 runs time backwards (flip/unflip).
// ---------------------------------------------------------------------------
__global__ __launch_bounds__(256)
void scan_kernel(const float* __restrict__ dt, const float* __restrict__ xdbl,
                 const float* __restrict__ xsact, const float* __restrict__ alog,
                 const float* __restrict__ dskip, const float* __restrict__ xz,
                 _Float16* __restrict__ y16, int reverse)
{
    const int gid = blockIdx.x * blockDim.x + threadIdx.x;
    if (gid >= NBATCH * D_INNER) return;
    const int d  = gid & (D_INNER - 1);
    const int bb = gid >> 9;
    float A[D_STATE], h[D_STATE];
#pragma unroll
    for (int s = 0; s < D_STATE; s++) { A[s] = -__expf(alog[d * D_STATE + s]); h[s] = 0.f; }
    const float dsk = dskip[d];
    for (int i = 0; i < SEQ; i++) {
        const int t = reverse ? (SEQ - 1 - i) : i;
        const size_t m   = ((size_t)bb << 12) + t;
        const float dtv  = dt[m * 512 + d];
        const float xv   = xsact[m * 512 + d];
        const float zv   = xz[m * 1024 + 512 + d];
        const float* Bc  = xdbl + m * 48 + 16;
        const float* Cc  = xdbl + m * 48 + 32;
        const float dx   = dtv * xv;
        float y = 0.f;
#pragma unroll
        for (int s = 0; s < D_STATE; s++) {
            h[s] = __expf(dtv * A[s]) * h[s] + dx * Bc[s];
            y += h[s] * Cc[s];
        }
        y += xv * dsk;
        const float gz = zv * (1.f / (1.f + __expf(-zv)));
        y16[m * 512 + d] = (_Float16)(y * gz);
    }
}

__global__ __launch_bounds__(256)
void cvt16_kernel(const float* __restrict__ s, _Float16* __restrict__ d, int n)
{
    const int i = blockIdx.x * blockDim.x + threadIdx.x;
    if (i < n) d[i] = (_Float16)s[i];
}

// ---------------------------------------------------------------------------
extern "C" void kernel_launch(void* const* d_in, const int* in_sizes, int n_in,
                              void* d_out, int out_size, void* d_ws, size_t ws_size,
                              hipStream_t stream)
{
    (void)in_sizes; (void)n_in; (void)out_size; (void)ws_size;
    const float* x      = (const float*)d_in[0];
    const float* fus_w  = (const float*)d_in[19];
    const float* fus_b  = (const float*)d_in[20];
    const float* ln1_g  = (const float*)d_in[21];
    const float* ln1_b  = (const float*)d_in[22];
    const float* ln2_g  = (const float*)d_in[23];
    const float* ln2_b  = (const float*)d_in[24];
    const float* mlp_w1 = (const float*)d_in[25];
    const float* mlp_b1 = (const float*)d_in[26];
    const float* mlp_w2 = (const float*)d_in[27];
    const float* mlp_b2 = (const float*)d_in[28];

    // ---- workspace carve-out (256B aligned; directions serialized & reuse) ----
    const uintptr_t base = (uintptr_t)d_ws;
    size_t off = 0;
    auto carve = [&](size_t bytes) -> void* {
        off = (off + 255) & ~(size_t)255;
        void* r = (void*)(base + off);
        off += bytes;
        return r;
    };
    const size_t M = MTOK;
    _Float16* h16     = (_Float16*)carve(M * 256 * 2);
    _Float16* win16   = (_Float16*)carve((size_t)1024 * 256 * 2);
    _Float16* wx16    = (_Float16*)carve((size_t)48 * 512 * 2);
    _Float16* wout16  = (_Float16*)carve((size_t)256 * 512 * 2);
    _Float16* fus16   = (_Float16*)carve((size_t)256 * 512 * 2);
    _Float16* m1w16   = (_Float16*)carve((size_t)1024 * 256 * 2);
    _Float16* m2w16   = (_Float16*)carve((size_t)256 * 1024 * 2);
    float*    xz      = (float*)carve(M * 1024 * 4);
    float*    xsact   = (float*)carve(M * 512 * 4);
    _Float16* xsact16 = (_Float16*)carve(M * 512 * 2);
    float*    xdbl    = (float*)carve(M * 48 * 4);
    float*    dtb     = (float*)carve(M * 512 * 4);
    _Float16* y16     = (_Float16*)carve(M * 512 * 2);
    _Float16* cat16   = (_Float16*)carve(M * 512 * 2);
    float*    x2      = (float*)carve(M * 256 * 4);
    _Float16* h2_16   = (_Float16*)carve(M * 256 * 2);
    _Float16* mlp116  = (_Float16*)carve(M * 1024 * 2);

    const dim3 blk(256);

    // LN1 -> h16 (f16 input to in_proj WMMA)
    ln_f16_kernel<<<M / 8, blk, 0, stream>>>(x, ln1_g, ln1_b, h16, (int)M);

    for (int dir = 0; dir < 2; dir++) {
        const int o = dir * 9;
        const float* win  = (const float*)d_in[1 + o];
        const float* cw   = (const float*)d_in[2 + o];
        const float* cb   = (const float*)d_in[3 + o];
        const float* wx   = (const float*)d_in[4 + o];
        const float* wdt  = (const float*)d_in[5 + o];
        const float* bdt  = (const float*)d_in[6 + o];
        const float* alog = (const float*)d_in[7 + o];
        const float* dskp = (const float*)d_in[8 + o];
        const float* wout = (const float*)d_in[9 + o];

        cvt16_kernel<<<(1024 * 256 + 255) / 256, blk, 0, stream>>>(win, win16, 1024 * 256);
        cvt16_kernel<<<(48 * 512 + 255) / 256, blk, 0, stream>>>(wx, wx16, 48 * 512);
        cvt16_kernel<<<(256 * 512 + 255) / 256, blk, 0, stream>>>(wout, wout16, 256 * 512);

        // in_proj: [M,256] x [1024,256]^T -> xz [M,1024] (f32)
        {
            const int waves = (int)(M / 32) * ((1024 / 16) / 4);   // MT=2, NT=4
            wmma_gemm_kernel<EPI_F32, 2, 4><<<waves / 8, blk, 0, stream>>>(
                h16, win16, (int)M, 1024, 256, xz, 1024, nullptr, 0, 0, nullptr, nullptr, 0);
        }
        // depthwise conv + SiLU (f32 + f16 copies)
        conv_silu_kernel<<<(int)(M * 512 / 256), blk, 0, stream>>>(xz, cw, cb, xsact, xsact16, dir);
        // x_proj: [M,512] x [48,512]^T -> xdbl [M,48]
        {
            const int waves = (int)(M / 16) * ((48 / 16) / 3);     // MT=1, NT=3
            wmma_gemm_kernel<EPI_F32, 1, 3><<<waves / 8, blk, 0, stream>>>(
                xsact16, wx16, (int)M, 48, 512, xdbl, 48, nullptr, 0, 0, nullptr, nullptr, 0);
        }
        // dt_proj + softplus
        dtproj_kernel<<<(int)(M * 512 / 256), blk, 0, stream>>>(xdbl, wdt, bdt, dtb);
        // selective scan (+ skip + silu(z) gating) -> y16 f16
        scan_kernel<<<(NBATCH * D_INNER) / 256, blk, 0, stream>>>(
            dtb, xdbl, xsact, alog, dskp, xz, y16, dir);
        // out_proj: [M,512] x [256,512]^T -> cat16[:, dir*256 .. +256) f16
        {
            const int waves = (int)(M / 32) * ((256 / 16) / 4);    // MT=2, NT=4
            wmma_gemm_kernel<EPI_F16, 2, 4><<<waves / 8, blk, 0, stream>>>(
                y16, wout16, (int)M, 256, 512, nullptr, 0, cat16, 512, dir * 256,
                nullptr, nullptr, 0);
        }
    }

    // fusion: x2 = x + cat @ fus_w^T + fus_b
    cvt16_kernel<<<(256 * 512 + 255) / 256, blk, 0, stream>>>(fus_w, fus16, 256 * 512);
    {
        const int waves = (int)(M / 32) * ((256 / 16) / 4);
        wmma_gemm_kernel<EPI_BIASRES, 2, 4><<<waves / 8, blk, 0, stream>>>(
            cat16, fus16, (int)M, 256, 512, x2, 256, nullptr, 0, 0, fus_b, x, 256);
    }
    // LN2 -> h2 (f16)
    ln_f16_kernel<<<M / 8, blk, 0, stream>>>(x2, ln2_g, ln2_b, h2_16, (int)M);
    // MLP1: silu(h2 @ w1^T + b1) -> f16
    cvt16_kernel<<<(1024 * 256 + 255) / 256, blk, 0, stream>>>(mlp_w1, m1w16, 1024 * 256);
    {
        const int waves = (int)(M / 32) * ((1024 / 16) / 4);
        wmma_gemm_kernel<EPI_SILU16, 2, 4><<<waves / 8, blk, 0, stream>>>(
            h2_16, m1w16, (int)M, 1024, 256, nullptr, 0, mlp116, 1024, 0,
            mlp_b1, nullptr, 0);
    }
    // MLP2: out = x2 + mlp1 @ w2^T + b2  -> d_out (f32 [M,256])
    cvt16_kernel<<<(256 * 1024 + 255) / 256, blk, 0, stream>>>(mlp_w2, m2w16, 256 * 1024);
    {
        const int waves = (int)(M / 32) * ((256 / 16) / 4);
        wmma_gemm_kernel<EPI_BIASRES, 2, 4><<<waves / 8, blk, 0, stream>>>(
            mlp116, m2w16, (int)M, 256, 1024, (float*)d_out, 256, nullptr, 0, 0,
            mlp_b2, x2, 256);
    }
}
